// DSVT_54090818126404
// MI455X (gfx1250) — compile-verified, matrix operand order checked by reading, not run on hardware
//
#include <hip/hip_runtime.h>
#include <hip/hip_bf16.h>
#include <cstdint>

typedef __bf16 bf16_t;
typedef __attribute__((ext_vector_type(16))) __bf16 v16bf;
typedef __attribute__((ext_vector_type(8)))  float  v8f;
typedef int v4i_t __attribute__((ext_vector_type(4)));

union FragB16 { v16bf v; unsigned int u[8]; };

#define N_VOX   262080
#define D_MODEL 192
#define N_HEAD  8
#define D_HEAD  24
#define D_FF    384
#define SET_SZ  36
#define N_SET   7280
#define N_LAYER 2

#if defined(__has_builtin)
#if __has_builtin(__builtin_amdgcn_global_load_async_to_lds_b128) && \
    __has_builtin(__builtin_amdgcn_s_wait_asynccnt)
#define HAVE_ASYNC_LDS 1
#endif
#endif
#ifndef HAVE_ASYNC_LDS
#define HAVE_ASYNC_LDS 0
#endif

#if HAVE_ASYNC_LDS
typedef __attribute__((address_space(1))) v4i_t* gas_v4p;
typedef __attribute__((address_space(3))) v4i_t* las_v4p;
// 16B global -> LDS async copy (ASYNCcnt-tracked, no VGPR round trip).
__device__ __forceinline__ void async_copy16(bf16_t* lds, const bf16_t* g) {
  __builtin_amdgcn_global_load_async_to_lds_b128(
      (gas_v4p)(uintptr_t)(const void*)g,
      (las_v4p)(uintptr_t)(void*)lds,
      0, 0);
}
#endif

// K index held by fragment VGPR v for the 16-bit 16x32 A layout (mirrored B layout):
// v=0..3 -> K = 2v + 8*hi ; v=4..7 -> K = 16 + 2(v-4) + 8*hi
__device__ __forceinline__ int frag_k(int v, int hi) {
  return ((v & 3) << 1) + ((v >> 2) << 4) + hi * 8;
}

__device__ __forceinline__ float gelu_tanh(float x) {
  float c = x + 0.044715f * x * x * x;
  return 0.5f * x * (1.0f + tanhf(0.7978845608028654f * c));
}

// ---------------------------------------------------------------- elementwise
__global__ __launch_bounds__(256)
void init_x_kernel(const float* __restrict__ src, float* __restrict__ x,
                   bf16_t* __restrict__ xb, long n) {
  long i = (long)blockIdx.x * 256 + threadIdx.x;
  if (i < n) { float v = src[i]; x[i] = v; xb[i] = (bf16_t)v; }
}

__global__ __launch_bounds__(256)
void add_pe_kernel(const float* __restrict__ x, const float* __restrict__ pe,
                   bf16_t* __restrict__ xpe, long n) {
  long i = (long)blockIdx.x * 256 + threadIdx.x;
  if (i < n) xpe[i] = (bf16_t)(x[i] + pe[i]);
}

__global__ __launch_bounds__(256)
void transpose_w_kernel(const float* __restrict__ W, bf16_t* __restrict__ Wt,
                        int Kd, int Nc) {  // W (Kd,Nc) -> Wt (Nc,Kd) bf16
  long i = (long)blockIdx.x * 256 + threadIdx.x;
  long tot = (long)Kd * Nc;
  if (i < tot) {
    int n = (int)(i / Kd), k = (int)(i % Kd);
    Wt[i] = (bf16_t)W[(long)k * Nc + n];
  }
}

__global__ __launch_bounds__(256)
void init_i32_kernel(int* __restrict__ p, int n, int val) {
  int i = blockIdx.x * 256 + threadIdx.x;
  if (i < n) p[i] = val;
}

__global__ __launch_bounds__(256)
void segmin_kernel(const int* __restrict__ inds, int* __restrict__ fp, int n) {
  int j = blockIdx.x * 256 + threadIdx.x;
  if (j < n) atomicMin(&fp[inds[j]], j);
}

// ---------------------------------------------------------------- WMMA GEMM
// C(M x NC) = A(M x KD) @ Wt^T + bias ; Wt stored (NC, KD) bf16.
// Block tile 64x192, 8 waves (2x4), wave tile 32x48 -> 6 wmma per 32-K step.
// A tile double-buffered in LDS via async global->LDS copies (ASYNCcnt).
template <bool GELU, bool OUTB, bool OUTF>
__global__ __launch_bounds__(256)
void gemm_bf16_wmma_kernel(const bf16_t* __restrict__ A,
                           const int* __restrict__ gather,
                           const bf16_t* __restrict__ Wt,
                           const float* __restrict__ bias,
                           bf16_t* __restrict__ outb,
                           float* __restrict__ outf,
                           int KD, int NC) {
  __shared__ bf16_t Alds[2][64 * 40];  // 80B row stride: conflict-free frag reads
  const int lane = threadIdx.x & 31;
  const int wave = threadIdx.x >> 5;
  const int wm = wave >> 2;            // 0..1
  const int wn = wave & 3;             // 0..3
  const int hi = lane >> 4;
  const int lo = lane & 15;
  const int mblk = blockIdx.x * 64;
  const int nblk = blockIdx.y * 192;

  v8f zf = {0.f, 0.f, 0.f, 0.f, 0.f, 0.f, 0.f, 0.f};
  v8f acc[2][3];
#pragma unroll
  for (int mt = 0; mt < 2; ++mt)
#pragma unroll
    for (int nt = 0; nt < 3; ++nt) acc[mt][nt] = zf;

  const int ldrow = threadIdx.x >> 2;  // 0..63
  const int ldq = threadIdx.x & 3;     // 0..3 (16B chunk)
  int grow = mblk + ldrow;
  long arow = gather ? (long)gather[grow] : (long)grow;
  const bf16_t* aptr = A + arow * (long)KD + ldq * 8;
  bf16_t* lds0 = &Alds[0][ldrow * 40 + ldq * 8];
  bf16_t* lds1 = &Alds[1][ldrow * 40 + ldq * 8];

  const int nsteps = KD >> 5;
#if HAVE_ASYNC_LDS
  async_copy16(lds0, aptr);  // prologue: tile 0 -> buf 0
#endif

  for (int s = 0; s < nsteps; ++s) {
    const int k0 = s << 5;
#if HAVE_ASYNC_LDS
    __builtin_amdgcn_s_wait_asynccnt(0);
    __syncthreads();  // all waves' tile-s data resident; prior reads of this buf done
    if (s + 1 < nsteps) async_copy16((s & 1) ? lds0 : lds1, aptr + k0 + 32);
#else
    __syncthreads();
    uint4 d = *(const uint4*)(aptr + k0);
    *(uint4*)((s & 1) ? lds1 : lds0) = d;
    __syncthreads();
#endif
    const bf16_t* buf = Alds[s & 1];

    FragB16 af[2];
#pragma unroll
    for (int mt = 0; mt < 2; ++mt) {
      int mrow = wm * 32 + mt * 16 + lo;
#pragma unroll
      for (int v = 0; v < 8; ++v) {
        int k = frag_k(v, hi);
        af[mt].u[v] = *(const unsigned int*)(&buf[mrow * 40 + k]);
      }
    }
#pragma unroll
    for (int nt = 0; nt < 3; ++nt) {
      FragB16 bfr;
      int ncol = nblk + wn * 48 + nt * 16 + lo;
      const bf16_t* wrow = Wt + (long)ncol * KD + k0;
#pragma unroll
      for (int v = 0; v < 8; ++v) {
        int k = frag_k(v, hi);
        bfr.u[v] = *(const unsigned int*)(wrow + k);
      }
#pragma unroll
      for (int mt = 0; mt < 2; ++mt)
        acc[mt][nt] = __builtin_amdgcn_wmma_f32_16x16x32_bf16(
            false, af[mt].v, false, bfr.v, (short)0, acc[mt][nt], false, false);
    }
  }

#pragma unroll
  for (int mt = 0; mt < 2; ++mt) {
    int rbase = mblk + wm * 32 + mt * 16 + hi * 8;
#pragma unroll
    for (int nt = 0; nt < 3; ++nt) {
      int col = nblk + wn * 48 + nt * 16 + lo;
      float bv = bias[col];
#pragma unroll
      for (int r = 0; r < 8; ++r) {
        float val = acc[mt][nt][r] + bv;
        if (GELU) val = gelu_tanh(val);
        long off = (long)(rbase + r) * NC + col;
        if (OUTF) outf[off] = val;
        if (OUTB) outb[off] = (bf16_t)val;
      }
    }
  }
}

// ---------------------------------------------------------------- attention
// One block per set (36 tokens), one wave per head. Padded 48x32 / 48x64 WMMA tiles.
// All fragment loads are branch-free: clamped in-bounds addresses + v_cndmask zeroing.
__global__ __launch_bounds__(256)
void attn_kernel(const bf16_t* __restrict__ Q, const bf16_t* __restrict__ K,
                 const bf16_t* __restrict__ V, const unsigned char* __restrict__ maskp,
                 bf16_t* __restrict__ O) {
  __shared__ bf16_t Vt[D_MODEL * SET_SZ];  // V transposed: Vt[c][tok]
  __shared__ bf16_t Pl[8 * 48 * 48];       // per-wave softmax matrix (bf16)
  const int set = blockIdx.x;
  const int tid = threadIdx.x;
  const int lane = tid & 31;
  const int h = tid >> 5;                  // wave == head
  const int hi = lane >> 4;
  const int lo = lane & 15;
  const long rowbase = (long)set * SET_SZ;

  for (int e = tid; e < SET_SZ * D_MODEL; e += 256) {
    int tok = e / D_MODEL, c = e - tok * D_MODEL;
    Vt[c * SET_SZ + tok] = V[(rowbase + tok) * D_MODEL + c];
  }
  __syncthreads();

  // hoisted, clamped key-padding-mask bytes (one per score column tile)
  unsigned char mflag[3];
#pragma unroll
  for (int nt = 0; nt < 3; ++nt) {
    int j = nt * 16 + lo;
    int jc = ((nt == 2) && (lo >= 4)) ? 35 : j;
    mflag[nt] = maskp[(long)set * SET_SZ + jc];
  }

  // ---- scores = Q Kt : A = Q rows (pad 48, K-dim pad 24->32), B = K rows as cols
  FragB16 bk[3];
#pragma unroll
  for (int nt = 0; nt < 3; ++nt) {
    int j = nt * 16 + lo;
    bool jv = (nt < 2) || (lo < 4);        // j < 36
    int jc = jv ? j : 35;
    const bf16_t* kp = K + (rowbase + jc) * D_MODEL + h * D_HEAD;
#pragma unroll
    for (int v = 0; v < 8; ++v) {
      int k = frag_k(v, hi);
      bool kv = (v < 4) || (hi == 0);      // k < 24
      int kc = kv ? k : 22;
      unsigned int u = *(const unsigned int*)(kp + kc);
      bk[nt].u[v] = (jv && kv) ? u : 0u;
    }
  }
  v8f zf = {0.f, 0.f, 0.f, 0.f, 0.f, 0.f, 0.f, 0.f};
  v8f sc[3][3];
#pragma unroll
  for (int mt = 0; mt < 3; ++mt)
#pragma unroll
    for (int nt = 0; nt < 3; ++nt) sc[mt][nt] = zf;

#pragma unroll
  for (int mt = 0; mt < 3; ++mt) {
    FragB16 aq;
    int m = mt * 16 + lo;
    bool mv = (mt < 2) || (lo < 4);        // m < 36
    int mc = mv ? m : 35;
    const bf16_t* qp = Q + (rowbase + mc) * D_MODEL + h * D_HEAD;
#pragma unroll
    for (int v = 0; v < 8; ++v) {
      int k = frag_k(v, hi);
      bool kv = (v < 4) || (hi == 0);      // k < 24
      int kc = kv ? k : 22;
      unsigned int u = *(const unsigned int*)(qp + kc);
      aq.u[v] = (mv && kv) ? u : 0u;
    }
#pragma unroll
    for (int nt = 0; nt < 3; ++nt)
      sc[mt][nt] = __builtin_amdgcn_wmma_f32_16x16x32_bf16(
          false, aq.v, false, bk[nt].v, (short)0, sc[mt][nt], false, false);
  }

  // ---- softmax (rows live in 16-lane halves; shfl_xor row reductions, wave32)
  const float scale = 0.20412414523193154f;  // 1/sqrt(24)
  bf16_t* Pw = Pl + h * 48 * 48;
#pragma unroll
  for (int mt = 0; mt < 3; ++mt) {
#pragma unroll
    for (int r = 0; r < 8; ++r) {
      int row = mt * 16 + hi * 8 + r;
      float v0[3];
      float mx = -3.0e38f;
#pragma unroll
      for (int nt = 0; nt < 3; ++nt) {
        bool oob = (nt == 2) && (lo >= 4);  // j >= 36
        float s = sc[mt][nt][r] * scale;
        s = mflag[nt] ? -1e9f : s;          // reference key_padding_mask semantics
        s = oob ? -3.0e38f : s;             // our padding: excluded entirely
        v0[nt] = s;
        mx = fmaxf(mx, s);
      }
#pragma unroll
      for (int d = 1; d < 16; d <<= 1) mx = fmaxf(mx, __shfl_xor(mx, d, 32));
      float sum = 0.f, e0[3];
#pragma unroll
      for (int nt = 0; nt < 3; ++nt) {
        bool oob = (nt == 2) && (lo >= 4);
        float e = oob ? 0.f : __expf(v0[nt] - mx);
        e0[nt] = e;
        sum += e;
      }
#pragma unroll
      for (int d = 1; d < 16; d <<= 1) sum += __shfl_xor(sum, d, 32);
      float inv = (sum > 0.f) ? (1.0f / sum) : 0.f;
#pragma unroll
      for (int nt = 0; nt < 3; ++nt)
        Pw[row * 48 + nt * 16 + lo] = (bf16_t)(e0[nt] * inv);
    }
  }

  // ---- O = P V : A = P (48 x 64 pad), B = V (tokens x dh) via transposed Vt
  FragB16 bv[2][2];
#pragma unroll
  for (int kc = 0; kc < 2; ++kc)
#pragma unroll
    for (int nt = 0; nt < 2; ++nt) {
      int n = nt * 16 + lo;
      bool nv = (nt == 0) || (lo < 8);     // n < 24
      int nc = nv ? n : 23;
      const bf16_t* vp = &Vt[(h * D_HEAD + nc) * SET_SZ];
#pragma unroll
      for (int v = 0; v < 8; ++v) {
        int kk = frag_k(v, hi) + kc * 32;
        bool kv = (kc == 0) || (v < 2 && hi == 0);  // kk < 36 (pair ends <= 35)
        int kkc = kv ? kk : 34;
        unsigned int u = *(const unsigned int*)(vp + kkc);
        bv[kc][nt].u[v] = (nv && kv) ? u : 0u;
      }
    }

  v8f oa[3][2];
#pragma unroll
  for (int mt = 0; mt < 3; ++mt)
#pragma unroll
    for (int nt = 0; nt < 2; ++nt) oa[mt][nt] = zf;

#pragma unroll
  for (int mt = 0; mt < 3; ++mt) {
#pragma unroll
    for (int kc = 0; kc < 2; ++kc) {
      FragB16 ap;
      int m = mt * 16 + lo;
#pragma unroll
      for (int v = 0; v < 8; ++v) {
        int kk = frag_k(v, hi) + kc * 32;
        unsigned int u = 0u;
        if (kc == 0 || v < 4)  // compile-time: kk <= 46 < 48 in-bounds, else zero tile
          u = *(const unsigned int*)(&Pw[m * 48 + kk]);
        ap.u[v] = u;
      }
#pragma unroll
      for (int nt = 0; nt < 2; ++nt)
        oa[mt][nt] = __builtin_amdgcn_wmma_f32_16x16x32_bf16(
            false, ap.v, false, bv[kc][nt].v, (short)0, oa[mt][nt], false, false);
    }
  }

  // ---- stage O tile in (now dead) per-wave P region, then coalesced store
  bf16_t* Ost = Pw;  // 48x32 u16 staging fits inside 48x48 region
#pragma unroll
  for (int mt = 0; mt < 3; ++mt)
#pragma unroll
    for (int nt = 0; nt < 2; ++nt)
#pragma unroll
      for (int r = 0; r < 8; ++r)
        Ost[(mt * 16 + hi * 8 + r) * 32 + nt * 16 + lo] = (bf16_t)oa[mt][nt][r];

  bf16_t* obase = O + rowbase * D_MODEL + h * D_HEAD;
  for (int e = lane; e < SET_SZ * 12; e += 32) {  // 12 dwords cover 24 bf16 per row
    int row = e / 12, cu = e - row * 12;
    unsigned int u = *(const unsigned int*)(&Ost[row * 32 + cu * 2]);
    *(unsigned int*)(obase + (long)row * D_MODEL + cu * 2) = u;
  }
}

// ---------------------------------------------------------------- residual + LayerNorm
// One wave per 192-wide row; optional gather of the residual (first_pos scatter-back).
__global__ __launch_bounds__(256)
void ln_residual_kernel(const float* __restrict__ x, const float* __restrict__ resid,
                        const int* __restrict__ gatherIdx,
                        const float* __restrict__ w, const float* __restrict__ b,
                        float* __restrict__ outx, bf16_t* __restrict__ outxb) {
  const int wave = threadIdx.x >> 5;
  const int lane = threadIdx.x & 31;
  const int row = blockIdx.x * 8 + wave;
  const long base = (long)row * D_MODEL;
  const long rbase = gatherIdx ? (long)gatherIdx[row] * D_MODEL : base;
  float vals[6];
  float s = 0.f;
#pragma unroll
  for (int i = 0; i < 6; ++i) {
    int c = lane + i * 32;
    float y = x[base + c] + resid[rbase + c];
    vals[i] = y;
    s += y;
  }
#pragma unroll
  for (int d = 1; d < 32; d <<= 1) s += __shfl_xor(s, d, 32);
  float mean = s * (1.0f / 192.0f);
  float vs = 0.f;
#pragma unroll
  for (int i = 0; i < 6; ++i) {
    float t = vals[i] - mean;
    vs += t * t;
  }
#pragma unroll
  for (int d = 1; d < 32; d <<= 1) vs += __shfl_xor(vs, d, 32);
  float rstd = rsqrtf(vs * (1.0f / 192.0f) + 1e-5f);
#pragma unroll
  for (int i = 0; i < 6; ++i) {
    int c = lane + i * 32;
    float o = (vals[i] - mean) * rstd * w[c] + b[c];
    outx[base + c] = o;
    if (outxb) outxb[base + c] = (bf16_t)o;
  }
}

// ---------------------------------------------------------------- launcher
extern "C" void kernel_launch(void* const* d_in, const int* in_sizes, int n_in,
                              void* d_out, int out_size, void* d_ws, size_t ws_size,
                              hipStream_t stream) {
  const float* src   = (const float*)d_in[0];
  const float* pos   = (const float*)d_in[1];
  const float* qkvw  = (const float*)d_in[2];
  const float* qkvb  = (const float*)d_in[3];
  const float* outw  = (const float*)d_in[4];
  const float* outbv = (const float*)d_in[5];
  const float* lnw   = (const float*)d_in[6];
  const float* lnb   = (const float*)d_in[7];
  const float* fw1   = (const float*)d_in[8];
  const float* fb1   = (const float*)d_in[9];
  const float* fw2   = (const float*)d_in[10];
  const float* fb2   = (const float*)d_in[11];
  const int*   svi   = (const int*)d_in[12];
  const unsigned char* svm = (const unsigned char*)d_in[13];

  char* ws = (char*)d_ws;
  const size_t SZ_XF = (size_t)N_VOX * D_MODEL * 4;  // 201,277,440
  const size_t SZ_XB = (size_t)N_VOX * D_MODEL * 2;  // 100,638,720
  float*  x   = (float*)(ws);
  bf16_t* xb  = (bf16_t*)(ws + SZ_XF);
  bf16_t* xpe = (bf16_t*)(ws + SZ_XF + 1 * SZ_XB);
  bf16_t* Qb  = (bf16_t*)(ws + SZ_XF + 2 * SZ_XB);
  bf16_t* Kb  = (bf16_t*)(ws + SZ_XF + 3 * SZ_XB);
  bf16_t* Vb  = (bf16_t*)(ws + SZ_XF + 4 * SZ_XB);
  bf16_t* Ob  = (bf16_t*)(ws + SZ_XF + 5 * SZ_XB);
  float*  oproj = (float*)Qb;       // reuse Q+K region after attention (2*SZ_XB)
  float*  ffout = (float*)Qb;       // reuse again after LN1 consumed oproj
  bf16_t* hbuf  = Vb;               // FFN hidden (N x 384 bf16) reuses V+O region
  int* fpos = (int*)(ws + SZ_XF + 6 * SZ_XB);
  char* wbase = ws + SZ_XF + 6 * SZ_XB + (size_t)N_VOX * 4;
  bf16_t* wtq = (bf16_t*)wbase;
  bf16_t* wtk = wtq + D_MODEL * D_MODEL;
  bf16_t* wtv = wtk + D_MODEL * D_MODEL;
  bf16_t* wto = wtv + D_MODEL * D_MODEL;
  bf16_t* wt1 = wto + D_MODEL * D_MODEL;   // (DFF, D)
  bf16_t* wt2 = wt1 + D_MODEL * D_FF;      // (D, DFF)

  const long nElem = (long)N_VOX * D_MODEL;
  const int eBlocks = (int)((nElem + 255) / 256);

  init_x_kernel<<<eBlocks, 256, 0, stream>>>(src, x, xb, nElem);

  for (int i = 0; i < N_LAYER; ++i) {
    const int* inds = svi + (size_t)i * N_VOX;
    const int wdd = (D_MODEL * D_MODEL + 255) / 256;
    const int wdf = (D_MODEL * D_FF + 255) / 256;
    transpose_w_kernel<<<wdd, 256, 0, stream>>>(qkvw + ((size_t)i * 3 + 0) * D_MODEL * D_MODEL, wtq, D_MODEL, D_MODEL);
    transpose_w_kernel<<<wdd, 256, 0, stream>>>(qkvw + ((size_t)i * 3 + 1) * D_MODEL * D_MODEL, wtk, D_MODEL, D_MODEL);
    transpose_w_kernel<<<wdd, 256, 0, stream>>>(qkvw + ((size_t)i * 3 + 2) * D_MODEL * D_MODEL, wtv, D_MODEL, D_MODEL);
    transpose_w_kernel<<<wdd, 256, 0, stream>>>(outw + (size_t)i * D_MODEL * D_MODEL, wto, D_MODEL, D_MODEL);
    transpose_w_kernel<<<wdf, 256, 0, stream>>>(fw1 + (size_t)i * D_MODEL * D_FF, wt1, D_MODEL, D_FF);
    transpose_w_kernel<<<wdf, 256, 0, stream>>>(fw2 + (size_t)i * D_FF * D_MODEL, wt2, D_FF, D_MODEL);

    add_pe_kernel<<<eBlocks, 256, 0, stream>>>(x, pos + (size_t)i * N_VOX * D_MODEL, xpe, nElem);

    init_i32_kernel<<<(N_VOX + 255) / 256, 256, 0, stream>>>(fpos, N_VOX, 0x7fffffff);
    segmin_kernel<<<(N_VOX + 255) / 256, 256, 0, stream>>>(inds, fpos, N_VOX);

    dim3 g1(N_VOX / 64, 1), g2(N_VOX / 64, 2);
    // Q/K from x+pos (gathered), V from x (gathered)
    gemm_bf16_wmma_kernel<false, true, false><<<g1, 256, 0, stream>>>(
        xpe, inds, wtq, qkvb + ((size_t)i * 3 + 0) * D_MODEL, Qb, nullptr, D_MODEL, D_MODEL);
    gemm_bf16_wmma_kernel<false, true, false><<<g1, 256, 0, stream>>>(
        xpe, inds, wtk, qkvb + ((size_t)i * 3 + 1) * D_MODEL, Kb, nullptr, D_MODEL, D_MODEL);
    gemm_bf16_wmma_kernel<false, true, false><<<g1, 256, 0, stream>>>(
        xb, inds, wtv, qkvb + ((size_t)i * 3 + 2) * D_MODEL, Vb, nullptr, D_MODEL, D_MODEL);

    attn_kernel<<<N_SET, 256, 0, stream>>>(Qb, Kb, Vb, svm + (size_t)i * N_VOX, Ob);

    gemm_bf16_wmma_kernel<false, false, true><<<g1, 256, 0, stream>>>(
        Ob, nullptr, wto, outbv + (size_t)i * D_MODEL, nullptr, oproj, D_MODEL, D_MODEL);

    ln_residual_kernel<<<N_VOX / 8, 256, 0, stream>>>(x, oproj, fpos,
        lnw + (size_t)(i * 2 + 0) * D_MODEL, lnb + (size_t)(i * 2 + 0) * D_MODEL, x, xb);

    gemm_bf16_wmma_kernel<true, true, false><<<g2, 256, 0, stream>>>(
        xb, nullptr, wt1, fb1 + (size_t)i * D_FF, hbuf, nullptr, D_MODEL, D_FF);
    gemm_bf16_wmma_kernel<false, false, true><<<g1, 256, 0, stream>>>(
        hbuf, nullptr, wt2, fb2 + (size_t)i * D_MODEL, nullptr, ffout, D_FF, D_MODEL);

    float*  xo  = (i == N_LAYER - 1) ? (float*)d_out : x;
    bf16_t* xbo = (i == N_LAYER - 1) ? nullptr : xb;
    ln_residual_kernel<<<N_VOX / 8, 256, 0, stream>>>(x, ffout, nullptr,
        lnw + (size_t)(i * 2 + 1) * D_MODEL, lnb + (size_t)(i * 2 + 1) * D_MODEL, xo, xbo);
  }
}
